// BlockV1_39230231282248
// MI455X (gfx1250) — compile-verified, matrix operand order checked by reading, not run on hardware
//
#include <hip/hip_runtime.h>

typedef unsigned short u16t;
typedef __attribute__((ext_vector_type(16))) __bf16 v16bf;
typedef __attribute__((ext_vector_type(8)))  float  v8f;
typedef __attribute__((address_space(3))) u16t lds_u16;

struct __align__(16) U4 { unsigned int x, y, z, w; };
union FragBF { v16bf v; U4 u[2]; };

static __device__ __forceinline__ u16t f2bf(float f) {
  unsigned int u = __float_as_uint(f);
  unsigned int r = (u + 0x7FFFu + ((u >> 16) & 1u)) >> 16;
  return (u16t)r;
}

static __device__ __forceinline__ float softplusf(float a) {
  // stable: max(a,0) + log1p(exp(-|a|))
  return fmaxf(a, 0.f) + log1pf(__expf(-fabsf(a)));
}

static __device__ __forceinline__ unsigned lds_off_u16(const u16t* p) {
  // __shared__ pointer -> 32-bit LDS byte offset (AS(3) pointers are 32-bit)
  return (unsigned)(size_t)(lds_u16*)p;
}

static __device__ __forceinline__ void async_a_load(unsigned lds_byte, const u16t* gp) {
  unsigned long long ga = (unsigned long long)(size_t)gp;
  // INST_OFFSET is added to both LDS and global addresses -> one base, two chunks
  asm volatile("global_load_async_to_lds_b128 %0, %1, off"
               :: "v"(lds_byte), "v"(ga) : "memory");
  asm volatile("global_load_async_to_lds_b128 %0, %1, off offset:16"
               :: "v"(lds_byte), "v"(ga) : "memory");
}

static __device__ __forceinline__ void wait_async0() {
  asm volatile("s_wait_asynccnt 0x0" ::: "memory");
}

// ---------------------------------------------------------------- casts
__global__ void cast_bf16_kernel(const float* __restrict__ in, u16t* __restrict__ out, int n) {
  int i = blockIdx.x * blockDim.x + threadIdx.x;
  if (i < n) out[i] = f2bf(in[i]);
}

// out[c*R + r] = in[r*C + c]  (transpose+cast; for pw_w [o,c] -> W[c][o])
__global__ void castT_bf16_kernel(const float* __restrict__ in, u16t* __restrict__ out, int R, int C) {
  int i = blockIdx.x * blockDim.x + threadIdx.x;
  if (i < R * C) {
    int r = i / C, c = i % C;
    out[(size_t)c * R + r] = f2bf(in[i]);
  }
}

// ------------------------------------------------ causal depthwise conv (K=4)
__global__ void conv_dw_kernel(const float* __restrict__ x, const float* __restrict__ w,
                               const float* __restrict__ bias, u16t* __restrict__ out,
                               int S, int D) {
  int idx = blockIdx.x * blockDim.x + threadIdx.x;  // over B*S*D (exact grid)
  int d = idx % D;
  int s = (idx / D) % S;
  int b = idx / (D * S);
  float acc = bias[d];
#pragma unroll
  for (int j = 0; j < 4; ++j) {
    int ss = s - 3 + j;
    if (ss >= 0) acc += x[((size_t)(b * S + ss)) * D + d] * w[d * 4 + j];
  }
  out[idx] = f2bf(acc);
}

// ------------------------------------------------------------- LayerNorm
// one wave (32 lanes) per row of D=512; writes fp32 and/or bf16 outputs
__global__ void ln_kernel(const float* __restrict__ x, const float* __restrict__ g,
                          const float* __restrict__ b, float* __restrict__ outF,
                          u16t* __restrict__ outB, int nrows) {
  const int D = 512;
  int row = blockIdx.x * 8 + (threadIdx.x >> 5);
  if (row >= nrows) return;
  int lane = threadIdx.x & 31;
  const float* xr = x + (size_t)row * D;
  float4 v[4];
  float s = 0.f, s2 = 0.f;
#pragma unroll
  for (int i = 0; i < 4; ++i) {
    v[i] = *(const float4*)(xr + i * 128 + lane * 4);
    s  += v[i].x + v[i].y + v[i].z + v[i].w;
    s2 += v[i].x * v[i].x + v[i].y * v[i].y + v[i].z * v[i].z + v[i].w * v[i].w;
  }
#pragma unroll
  for (int off = 16; off > 0; off >>= 1) {
    s  += __shfl_xor(s,  off, 32);
    s2 += __shfl_xor(s2, off, 32);
  }
  float mean = s * (1.f / 512.f);
  float var  = s2 * (1.f / 512.f) - mean * mean;
  float rstd = rsqrtf(var + 1e-5f);
#pragma unroll
  for (int i = 0; i < 4; ++i) {
    int c0 = i * 128 + lane * 4;
    float4 gv = *(const float4*)(g + c0);
    float4 bv = *(const float4*)(b + c0);
    float4 o;
    o.x = (v[i].x - mean) * rstd * gv.x + bv.x;
    o.y = (v[i].y - mean) * rstd * gv.y + bv.y;
    o.z = (v[i].z - mean) * rstd * gv.z + bv.z;
    o.w = (v[i].w - mean) * rstd * gv.w + bv.w;
    if (outF) *(float4*)(outF + (size_t)row * D + c0) = o;
    if (outB) {
      u16t* p = outB + (size_t)row * D + c0;
      p[0] = f2bf(o.x); p[1] = f2bf(o.y); p[2] = f2bf(o.z); p[3] = f2bf(o.w);
    }
  }
}

// ------------------------------------------------------- log-space GRU scan
// thread per (b,d); L_s = logaddexp(L_{s-1} + log_c[s], log_z[s]+log_th[s])
__global__ void scan_kernel(const float* __restrict__ kz, const float* __restrict__ kh,
                            float* __restrict__ x, int B, int S, int D) {
  int t = blockIdx.x * blockDim.x + threadIdx.x;
  if (t >= B * D) return;
  int b = t / D, d = t % D;
  size_t base = (size_t)b * S * D + d;
  float logh = -0.69314718056f;  // log 0.5 = log_g(0)
  for (int s = 0; s < S; ++s) {
    size_t off = base + (size_t)s * D;
    float kzv = kz[off];
    float khv = kh[off];
    float log_z = -softplusf(-kzv);
    float log_c = -softplusf(kzv);
    float log_th = (khv >= 0.f) ? __logf(khv + 0.5f) : -softplusf(-khv);
    float lv = log_z + log_th;
    float la = log_c + logh;
    float mx = fmaxf(la, lv);
    logh = mx + log1pf(__expf(fminf(la, lv) - mx));
    x[off] += __expf(logh);
  }
}

// --------------------------------------------------------- WMMA bf16 GEMM
// C[N x Dout] = act(A_bf16[N x Kt] @ W_bf16[Kt x Dout] + bias) (+ res)
// block: 256 threads (8 waves), 128x128 tile; wave: 32x64 via 2x4 wmma tiles.
// Double-buffered LDS; A tile streamed via global_load_async_to_lds_b128
// (ASYNCcnt), B tile transposed through registers (K-major in LDS).
#define LDT 40  // LDS row stride in u16 (80 B, multiple of 16 B)
template <bool BIAS, bool RELU, bool RES, bool OUTF, bool OUTB>
__global__ __launch_bounds__(256) void gemm_bf16_kernel(
    const u16t* __restrict__ A, const u16t* __restrict__ Wg,
    const float* __restrict__ bias, const float* __restrict__ res,
    float* __restrict__ Cf, u16t* __restrict__ Cb,
    int Kt, int Dout) {
  __shared__ __align__(16) u16t As[2][128 * LDT];
  __shared__ __align__(16) u16t Bs[2][128 * LDT];

  const int tid  = threadIdx.x;
  const int lane = tid & 31;
  const int wave = tid >> 5;
  const int wr = wave >> 1;  // 0..3 : 32-row block
  const int wc = wave & 1;   // 0..1 : 64-col block
  const int row0 = blockIdx.y * 128;
  const int col0 = blockIdx.x * 128;

  // A-tile: thread -> (row 0..127, 16-elem segment 0..1); 32 B via 2 async b128
  const int ar = tid >> 1;
  const int aseg = (tid & 1) * 16;
  const unsigned ldsA[2] = { lds_off_u16(&As[0][ar * LDT + aseg]),
                             lds_off_u16(&As[1][ar * LDT + aseg]) };
  // B-tile: thread -> (col 0..127, K chunks {h*8, h*8+16}); K-major in LDS
  const int bc = tid & 127;
  const int bks = (tid >> 7) * 8;  // 0 or 8

  const int g = lane >> 4;     // K-half selector per ISA layout
  const int l16 = lane & 15;

  v8f acc[2][4];
#pragma unroll
  for (int m = 0; m < 2; ++m)
#pragma unroll
    for (int n = 0; n < 4; ++n)
#pragma unroll
      for (int i = 0; i < 8; ++i) acc[m][n][i] = 0.f;

  const int nk = Kt >> 5;

  // ---- prologue: stage tile 0 into buffer 0
  {
    const u16t* ap = A + (size_t)(row0 + ar) * Kt + aseg;
    async_a_load(ldsA[0], ap);
    __align__(16) u16t bt0[8], bt1[8];
#pragma unroll
    for (int i = 0; i < 8; ++i) bt0[i] = Wg[(size_t)(bks + i) * Dout + col0 + bc];
#pragma unroll
    for (int i = 0; i < 8; ++i) bt1[i] = Wg[(size_t)(bks + 16 + i) * Dout + col0 + bc];
    *(U4*)&Bs[0][bc * LDT + bks]      = *(const U4*)bt0;
    *(U4*)&Bs[0][bc * LDT + bks + 16] = *(const U4*)bt1;
    wait_async0();
    __syncthreads();
  }

  for (int kt = 0; kt < nk; ++kt) {
    const int cur = kt & 1;
    const bool more = (kt + 1) < nk;

    // ---- prefetch tile kt+1 into the other buffer (overlaps WMMAs below)
    __align__(16) u16t bt0[8], bt1[8];
    if (more) {
      const int k0 = (kt + 1) << 5;
      const u16t* ap = A + (size_t)(row0 + ar) * Kt + k0 + aseg;
      async_a_load(ldsA[cur ^ 1], ap);
#pragma unroll
      for (int i = 0; i < 8; ++i) bt0[i] = Wg[(size_t)(k0 + bks + i) * Dout + col0 + bc];
#pragma unroll
      for (int i = 0; i < 8; ++i) bt1[i] = Wg[(size_t)(k0 + bks + 16 + i) * Dout + col0 + bc];
    }

    // ---- compute on current buffer
    // fragment gather per ISA 16-bit layout: lane L -> row/col L%16,
    // K chunks [8g, 8g+8) and [16+8g, 16+8g+8)
    FragBF fa[2], fb[4];
#pragma unroll
    for (int m = 0; m < 2; ++m) {
      int r = wr * 32 + m * 16 + l16;
      fa[m].u[0] = *(const U4*)&As[cur][r * LDT + g * 8];
      fa[m].u[1] = *(const U4*)&As[cur][r * LDT + 16 + g * 8];
    }
#pragma unroll
    for (int n = 0; n < 4; ++n) {
      int c = wc * 64 + n * 16 + l16;
      fb[n].u[0] = *(const U4*)&Bs[cur][c * LDT + g * 8];
      fb[n].u[1] = *(const U4*)&Bs[cur][c * LDT + 16 + g * 8];
    }
#pragma unroll
    for (int m = 0; m < 2; ++m)
#pragma unroll
      for (int n = 0; n < 4; ++n)
        acc[m][n] = __builtin_amdgcn_wmma_f32_16x16x32_bf16(
            false, fa[m].v, false, fb[n].v, (short)0, acc[m][n], false, false);

    if (more) {
      *(U4*)&Bs[cur ^ 1][bc * LDT + bks]      = *(const U4*)bt0;
      *(U4*)&Bs[cur ^ 1][bc * LDT + bks + 16] = *(const U4*)bt1;
      wait_async0();    // own async A writes to next buffer done
      __syncthreads();  // everyone's stores visible; cur-buffer reads done
    }
  }

  // ---- epilogue: bias / relu / residual / fp32 and-or bf16 store (branch-free)
#pragma unroll
  for (int m = 0; m < 2; ++m) {
#pragma unroll
    for (int n = 0; n < 4; ++n) {
      int colg = col0 + wc * 64 + n * 16 + l16;
      float bv = BIAS ? bias[colg] : 0.f;
#pragma unroll
      for (int v = 0; v < 8; ++v) {
        int rowg = row0 + wr * 32 + m * 16 + v + 8 * g;
        float val = acc[m][n][v] + bv;
        if (RELU) val = fmaxf(val, 0.f);
        size_t off = (size_t)rowg * Dout + colg;
        if (RES) val += res[off];
        if (OUTF) Cf[off] = val;
        if (OUTB) Cb[off] = f2bf(val);
      }
    }
  }
}

// ---------------------------------------------------------------- driver
extern "C" void kernel_launch(void* const* d_in, const int* in_sizes, int n_in,
                              void* d_out, int out_size, void* d_ws, size_t ws_size,
                              hipStream_t stream) {
  (void)in_sizes; (void)n_in; (void)out_size; (void)ws_size;
  const int B = 4, S = 4096, D = 512, H = 2048, L = 3;
  const int N = B * S;  // 16384 rows

  const float* x     = (const float*)d_in[0];
  const float* dw_w  = (const float*)d_in[1];
  const float* dw_b  = (const float*)d_in[2];
  const float* pw_w  = (const float*)d_in[3];
  const float* pw_b  = (const float*)d_in[4];
  const float* ln1_g = (const float*)d_in[5];
  const float* ln1_b = (const float*)d_in[6];
  const float* Wz    = (const float*)d_in[7];
  const float* Wh    = (const float*)d_in[8];
  const float* ln_g  = (const float*)d_in[9];
  const float* ln_b  = (const float*)d_in[10];
  const float* ln2_g = (const float*)d_in[11];
  const float* ln2_b = (const float*)d_in[12];
  const float* W1    = (const float*)d_in[13];
  const float* b1    = (const float*)d_in[14];
  const float* W2    = (const float*)d_in[15];
  const float* b2    = (const float*)d_in[16];
  float* out = (float*)d_out;

  char* wsp = (char*)d_ws;
  size_t o = 0;
  auto walloc = [&](size_t bytes) -> char* {
    char* p = wsp + o;
    o += (bytes + 255) & ~(size_t)255;
    return p;
  };
  u16t*  convB = (u16t*)walloc((size_t)N * D * 2);
  float* xcur  = (float*)walloc((size_t)N * D * 4);
  float* gruF  = (float*)walloc((size_t)N * D * 4);
  u16t*  gruB  = (u16t*)walloc((size_t)N * D * 2);
  float* kz    = (float*)walloc((size_t)N * D * 4);
  float* kh    = (float*)walloc((size_t)N * D * 4);
  u16t*  mlpB  = (u16t*)walloc((size_t)N * D * 2);
  u16t*  h1B   = (u16t*)walloc((size_t)N * H * 2);
  u16t*  pwT   = (u16t*)walloc((size_t)D * D * 2);
  u16t*  WzB   = (u16t*)walloc((size_t)L * D * D * 2);
  u16t*  WhB   = (u16t*)walloc((size_t)L * D * D * 2);
  u16t*  W1B   = (u16t*)walloc((size_t)D * H * 2);
  u16t*  W2B   = (u16t*)walloc((size_t)H * D * 2);

  // weight casts (bf16; pw_w transposed so GEMM computes y @ pw_w^T)
  castT_bf16_kernel<<<(D * D + 255) / 256, 256, 0, stream>>>(pw_w, pwT, D, D);
  cast_bf16_kernel<<<(L * D * D + 255) / 256, 256, 0, stream>>>(Wz, WzB, L * D * D);
  cast_bf16_kernel<<<(L * D * D + 255) / 256, 256, 0, stream>>>(Wh, WhB, L * D * D);
  cast_bf16_kernel<<<(D * H + 255) / 256, 256, 0, stream>>>(W1, W1B, D * H);
  cast_bf16_kernel<<<(H * D + 255) / 256, 256, 0, stream>>>(W2, W2B, H * D);

  // causal depthwise conv -> bf16
  conv_dw_kernel<<<(N * D + 255) / 256, 256, 0, stream>>>(x, dw_w, dw_b, convB, S, D);

  // pointwise conv + bias + residual(x) -> xcur
  gemm_bf16_kernel<true, false, true, true, false>
      <<<dim3(D / 128, N / 128), 256, 0, stream>>>(
          convB, pwT, pw_b, x, xcur, nullptr, D, D);

  // LN1 -> gru_in (fp32 + bf16)
  ln_kernel<<<(N + 7) / 8, 256, 0, stream>>>(xcur, ln1_g, ln1_b, gruF, gruB, N);

  for (int i = 0; i < L; ++i) {
    gemm_bf16_kernel<false, false, false, true, false>
        <<<dim3(D / 128, N / 128), 256, 0, stream>>>(
            gruB, WzB + (size_t)i * D * D, nullptr, nullptr, kz, nullptr, D, D);
    gemm_bf16_kernel<false, false, false, true, false>
        <<<dim3(D / 128, N / 128), 256, 0, stream>>>(
            gruB, WhB + (size_t)i * D * D, nullptr, nullptr, kh, nullptr, D, D);
    // xcur += h (in-place: each element read once by its owning thread)
    scan_kernel<<<(B * D + 255) / 256, 256, 0, stream>>>(kz, kh, xcur, B, S, D);
    ln_kernel<<<(N + 7) / 8, 256, 0, stream>>>(
        xcur, ln_g + (size_t)i * D, ln_b + (size_t)i * D, gruF, gruB, N);
  }

  // LN2 on gru_in -> mlp_in (bf16 only)
  ln_kernel<<<(N + 7) / 8, 256, 0, stream>>>(gruF, ln2_g, ln2_b, nullptr, mlpB, N);

  // MLP: relu(mlp_in @ W1 + b1) -> bf16 ; @ W2 + b2 + xcur -> d_out
  gemm_bf16_kernel<true, true, false, false, true>
      <<<dim3(H / 128, N / 128), 256, 0, stream>>>(
          mlpB, W1B, b1, nullptr, nullptr, h1B, D, H);
  gemm_bf16_kernel<true, false, true, true, false>
      <<<dim3(D / 128, N / 128), 256, 0, stream>>>(
          h1B, W2B, b2, xcur, out, nullptr, H, D);
}
